// EncoderLayer_3272765079829
// MI455X (gfx1250) — matrix operand order repaired and checked by reading, not compiled, not run on hardware
//
#include <hip/hip_runtime.h>
#include <cstdint>
#include <cstddef>

// ---------------------------------------------------------------------------
// Types for CDNA5 WMMA (gfx1250, wave32)
// ---------------------------------------------------------------------------
typedef __bf16 bf16;
typedef bf16  v16bf __attribute__((ext_vector_type(16)));
typedef float v8f   __attribute__((ext_vector_type(8)));
typedef int   v4i   __attribute__((ext_vector_type(4)));
typedef unsigned short u16;

struct PairV4 { v4i lo; v4i hi; };

__device__ __forceinline__ v16bf mkfrag(v4i lo, v4i hi) {
    PairV4 p{lo, hi};
    return __builtin_bit_cast(v16bf, p);
}

__device__ __forceinline__ v8f wmma_bf16(v16bf a, v16bf b, v8f c) {
    // D = A(16x32 bf16) * B(32x16 bf16) + C(16x16 f32)
    return __builtin_amdgcn_wmma_f32_16x16x32_bf16(
        /*neg_a=*/false, a, /*neg_b=*/false, b,
        /*c_mod=*/(short)0, c, /*reuse_a=*/false, /*reuse_b=*/false);
}

__device__ __forceinline__ u16 f2bf(float f) {
    unsigned int u = __builtin_bit_cast(unsigned int, f);
    u += 0x7FFFu + ((u >> 16) & 1u);   // round-to-nearest-even
    return (u16)(u >> 16);
}

__device__ __forceinline__ float gelu_exact(float x) {
    return 0.5f * x * (1.0f + erff(x * 0.70710678118654752f));
}

// ---------------------------------------------------------------------------
// CDNA5 async global->LDS staging (ASYNCcnt) with compile-safe fallback.
// Probe result: builtin exists; param types are v4i in AS(1)/AS(3).
// ---------------------------------------------------------------------------
#if defined(__AMDGCN__) && __has_builtin(__builtin_amdgcn_global_load_async_to_lds_b128)
#define HAS_ASYNC 1
typedef __attribute__((address_space(1))) v4i GV4;
typedef __attribute__((address_space(3))) v4i LV4;
__device__ __forceinline__ void async_b128(const void* g, void* l) {
    __builtin_amdgcn_global_load_async_to_lds_b128((GV4*)g, (LV4*)l,
                                                   /*offset=*/0, /*cpol=*/0);
}
__device__ __forceinline__ void wait_async0() {
#if __has_builtin(__builtin_amdgcn_s_wait_asynccnt)
    __builtin_amdgcn_s_wait_asynccnt(0);
#else
    asm volatile("s_wait_asynccnt 0" ::: "memory");
#endif
}
#else
#define HAS_ASYNC 0
__device__ __forceinline__ void async_b128(const void*, void*) {}
__device__ __forceinline__ void wait_async0() {}
#endif

// Problem constants
#define BB 2
#define LL 2048
#define DD 1024
#define NH 16
#define HDIM 64
#define MM (BB * LL)     // 4096 rows
#define FF (4 * DD)      // 4096

// ---------------------------------------------------------------------------
// fp32 -> bf16 conversion, 8 elements / thread (b128 in, b128 out)
// ---------------------------------------------------------------------------
__global__ __launch_bounds__(256) void cvt_kernel(const float* __restrict__ in,
                                                  u16* __restrict__ outp, int n) {
    int i = (blockIdx.x * 256 + threadIdx.x) * 8;
    if (i >= n) return;
    union { v4i v; u16 s[8]; } o;
#pragma unroll
    for (int j = 0; j < 8; ++j) o.s[j] = f2bf(in[i + j]);
    *(v4i*)(outp + i) = o.v;
}

// ---------------------------------------------------------------------------
// Tiled WMMA GEMM: C[M,Nc] = A[M,K](bf16) * B[K,Nc](bf16)
// Block 256 thr = 8 waves, tile 128x128, K-step 32, double-buffered LDS.
// A tile staged with global_load_async_to_lds_b128 (ping-pong, ASYNCcnt);
// B tile prefetched to registers, transpose-packed to LDS after the WMMAs.
// STORE: 0 = f32 plain, 1 = bf16 plain, 2 = bf16 head-scatter ([b,n,l,hd])
// ---------------------------------------------------------------------------
template <bool BIAS, bool GELU, int STORE>
__global__ __launch_bounds__(256) void gemm_wmma_kernel(
    const u16* __restrict__ A, const u16* __restrict__ Bm,
    const float* __restrict__ bias, void* __restrict__ outp,
    int M, int K, int Nc, float outscale) {

    __shared__ u16 As[2][128][40];   // row stride 80B (16B aligned, conflict-free)
    __shared__ u16 Bs[2][128][40];   // B tile stored transposed: Bs[.][n][k]

    const int tid  = threadIdx.x;
    const int lane = tid & 31;
    const int w    = tid >> 5;       // 0..7
    const int wm   = w >> 1;         // 0..3  -> +32*wm rows
    const int wn   = w & 1;          // 0..1  -> +64*wn cols
    const int lr   = lane & 15;
    const int hi   = (lane < 16) ? 0 : 1;

    const int m0 = blockIdx.y * 128;
    const int n0 = blockIdx.x * 128;

    // staging roles
    const int arow = tid >> 1, ahk = (tid & 1) * 16;   // A: row, k-half
    const int kp   = tid >> 4, ng  = tid & 15;         // B: k-pair, n-group

    union U8 { v4i v; u16 s[8]; };
    U8 br0, br1;   // B prefetch registers (live across the WMMA block)

    auto issueA = [&](int buf, int k0) {
        const u16* pA = A + (size_t)(m0 + arow) * K + k0 + ahk;
#if HAS_ASYNC
        async_b128(pA,     &As[buf][arow][ahk]);
        async_b128(pA + 8, &As[buf][arow][ahk + 8]);
#else
        v4i a0 = *(const v4i*)(pA);
        v4i a1 = *(const v4i*)(pA + 8);
        *(v4i*)&As[buf][arow][ahk]     = a0;
        *(v4i*)&As[buf][arow][ahk + 8] = a1;
#endif
    };
    auto loadB = [&](int k0) {
        const u16* pB = Bm + (size_t)(k0 + 2 * kp) * Nc + n0 + 8 * ng;
        br0.v = *(const v4i*)(pB);
        br1.v = *(const v4i*)(pB + Nc);
    };
    auto storeB = [&](int buf) {
#pragma unroll
        for (int j = 0; j < 8; ++j) {
            unsigned int pk = (unsigned int)br0.s[j] | ((unsigned int)br1.s[j] << 16);
            *(unsigned int*)&Bs[buf][8 * ng + j][2 * kp] = pk;
        }
    };

    v8f acc[2][4];
#pragma unroll
    for (int i = 0; i < 2; ++i)
#pragma unroll
        for (int j = 0; j < 4; ++j) acc[i][j] = {};

    // prologue: fully stage tile 0
    issueA(0, 0);
    loadB(0);
    storeB(0);

    const int nk = K >> 5;
    for (int kt = 0; kt < nk; ++kt) {
        const int cur = kt & 1, nxt = cur ^ 1;
        wait_async0();          // complete this wave's async A stores to LDS
        __syncthreads();        // all waves' tile data visible; prev compute done

        if (kt + 1 < nk) {      // kick off next tile while we do the math
            issueA(nxt, 32 * (kt + 1));
            loadB(32 * (kt + 1));
        }

        // ---- fragments + 8 WMMAs for current tile ----
        v16bf af[2];
#pragma unroll
        for (int mf = 0; mf < 2; ++mf) {
            const u16* pr = &As[cur][wm * 32 + mf * 16 + lr][hi * 16];
            af[mf] = mkfrag(*(const v4i*)pr, *(const v4i*)(pr + 8));
        }
#pragma unroll
        for (int nf = 0; nf < 4; ++nf) {
            const u16* pb = &Bs[cur][wn * 64 + nf * 16 + lr][hi * 16];
            v16bf bf = mkfrag(*(const v4i*)pb, *(const v4i*)(pb + 8));
#pragma unroll
            for (int mf = 0; mf < 2; ++mf)
                acc[mf][nf] = wmma_bf16(af[mf], bf, acc[mf][nf]);
        }

        if (kt + 1 < nk) storeB(nxt);   // transpose-pack next B after the math
    }

    // ---- epilogue ----
    float* of = (float*)outp;
    u16*   ob = (u16*)outp;
#pragma unroll
    for (int nf = 0; nf < 4; ++nf) {
        const int col = n0 + wn * 64 + nf * 16 + lr;
        float bcol = 0.0f;
        if (BIAS) bcol = bias[col];
#pragma unroll
        for (int mf = 0; mf < 2; ++mf) {
#pragma unroll
            for (int r = 0; r < 8; ++r) {
                const int row = m0 + wm * 32 + mf * 16 + r + hi * 8;
                float v = acc[mf][nf][r];
                if (BIAS) v += bcol;
                if (GELU) v = gelu_exact(v);
                v *= outscale;
                if (STORE == 0) {
                    of[(size_t)row * Nc + col] = v;
                } else if (STORE == 1) {
                    ob[(size_t)row * Nc + col] = f2bf(v);
                } else {
                    // head scatter: row=(b,l), col=d=(hd*16+n) -> [b][n][l][hd]
                    const int b  = row >> 11, l = row & 2047;
                    const int nh = col & 15,  hd = col >> 4;
                    ob[(((size_t)b * NH + nh) * LL + l) * HDIM + hd] = f2bf(v);
                }
            }
        }
    }
}

// ---------------------------------------------------------------------------
// Flash attention (V := K, faithful to reference). One block = (b,head) x 64
// q-rows; 4 waves x 16 rows. Keys streamed in 32-wide tiles through LDS.
// Qh/Kh layout: [b][n][l][hd] bf16, Q pre-scaled by 1/8.
// ---------------------------------------------------------------------------
__global__ __launch_bounds__(128) void flash_kernel(const u16* __restrict__ Qh,
                                                    const u16* __restrict__ Kh,
                                                    u16* __restrict__ ctx) {
    __shared__ u16 Krow[32][72];        // [m][hd], stride 144B
    __shared__ u16 Ktr[64][40];         // [hd][m], stride 80B
    __shared__ u16 Pbuf[4][16][40];     // per-wave softmax-weight staging

    const int tid  = threadIdx.x;
    const int lane = tid & 31;
    const int w    = tid >> 5;          // 0..3
    const int lr   = lane & 15;
    const int hi   = (lane < 16) ? 0 : 1;
    const int bn   = blockIdx.y;        // b*16 + head
    const int b    = bn >> 4, nh = bn & 15;
    const size_t base = (size_t)bn * LL * HDIM;

    // Q A-fragments for this wave's 16 rows (two k-steps over hd=64)
    v16bf qa[2];
    {
        const int qrow = blockIdx.x * 64 + w * 16 + lr;
        const u16* qp = Qh + base + (size_t)qrow * HDIM;
#pragma unroll
        for (int ks = 0; ks < 2; ++ks) {
            const int c0 = ks * 32 + hi * 16;
            qa[ks] = mkfrag(*(const v4i*)(qp + c0), *(const v4i*)(qp + c0 + 8));
        }
    }

    v8f o[4];
#pragma unroll
    for (int i = 0; i < 4; ++i) o[i] = {};
    float rm[8], rs[8];
#pragma unroll
    for (int r = 0; r < 8; ++r) { rm[r] = -1e30f; rs[r] = 0.0f; }

    for (int m0 = 0; m0 < LL; m0 += 32) {
        __syncthreads();
        // ---- stage 32x64 key tile, both layouts ----
        {
            const int m = tid >> 2, seg = tid & 3;
            const u16* kp = Kh + base + (size_t)(m0 + m) * HDIM + seg * 16;
            union { v4i v; u16 s[8]; } ua, ub;
            ua.v = *(const v4i*)(kp);
            ub.v = *(const v4i*)(kp + 8);
            *(v4i*)&Krow[m][seg * 16]     = ua.v;
            *(v4i*)&Krow[m][seg * 16 + 8] = ub.v;
#pragma unroll
            for (int j = 0; j < 8; ++j) {
                Ktr[seg * 16 + j][m]     = ua.s[j];
                Ktr[seg * 16 + 8 + j][m] = ub.s[j];
            }
        }
        __syncthreads();

        // ---- S = Q * K^T  (16 x 32 per wave) ----
        v8f s0 = {}, s1 = {};
#pragma unroll
        for (int ks = 0; ks < 2; ++ks) {
            const int kh = ks * 32 + hi * 16;
            v16bf b0 = mkfrag(*(const v4i*)&Krow[lr][kh],      *(const v4i*)&Krow[lr][kh + 8]);
            v16bf b1 = mkfrag(*(const v4i*)&Krow[16 + lr][kh], *(const v4i*)&Krow[16 + lr][kh + 8]);
            s0 = wmma_bf16(qa[ks], b0, s0);
            s1 = wmma_bf16(qa[ks], b1, s1);
        }

        // ---- online softmax (fp32), stage P as bf16 A-fragments ----
#pragma unroll
        for (int r = 0; r < 8; ++r) {
            float a0 = s0[r], a1 = s1[r];
            float mx = fmaxf(a0, a1);
#pragma unroll
            for (int off = 8; off; off >>= 1) mx = fmaxf(mx, __shfl_xor(mx, off, 16));
            const float mnew  = fmaxf(rm[r], mx);
            const float scale = expf(rm[r] - mnew);
            const float p0 = expf(a0 - mnew);
            const float p1 = expf(a1 - mnew);
            float sum = p0 + p1;
#pragma unroll
            for (int off = 8; off; off >>= 1) sum += __shfl_xor(sum, off, 16);
            rs[r] = rs[r] * scale + sum;
            rm[r] = mnew;
#pragma unroll
            for (int hs = 0; hs < 4; ++hs) o[hs][r] *= scale;
            const int lrow = r + hi * 8;
            Pbuf[w][lrow][lr]      = f2bf(p0);
            Pbuf[w][lrow][16 + lr] = f2bf(p1);
        }
        __builtin_amdgcn_wave_barrier();   // wave-local LDS round-trip

        // ---- O += P(16x32) * K(32x64) ----
        const int c0 = hi * 16;
        v16bf pf = mkfrag(*(const v4i*)&Pbuf[w][lr][c0], *(const v4i*)&Pbuf[w][lr][c0 + 8]);
        const int kh = hi * 16;
#pragma unroll
        for (int hs = 0; hs < 4; ++hs) {
            const u16* pb = &Ktr[hs * 16 + lr][kh];
            v16bf bf = mkfrag(*(const v4i*)pb, *(const v4i*)(pb + 8));
            o[hs] = wmma_bf16(pf, bf, o[hs]);
        }
    }

    // ---- normalize & scatter ctx back to [b][l][d], d = hd*16 + n ----
#pragma unroll
    for (int hs = 0; hs < 4; ++hs) {
#pragma unroll
        for (int r = 0; r < 8; ++r) {
            const int l = blockIdx.x * 64 + w * 16 + r + hi * 8;
            const int d = (hs * 16 + lr) * NH + nh;
            const float v = o[hs][r] / rs[r];
            ctx[((size_t)b * LL + l) * DD + d] = f2bf(v);
        }
    }
}

// ---------------------------------------------------------------------------
// h = LayerNorm(xa + xb) * gamma + beta ; writes fp32 (+ optional bf16 copy)
// ---------------------------------------------------------------------------
__global__ __launch_bounds__(256) void add_ln_kernel(
    const float* __restrict__ xa, const float* __restrict__ xb,
    const float* __restrict__ gamma, const float* __restrict__ beta,
    float* __restrict__ o32, u16* __restrict__ o16) {

    __shared__ float red[256];
    __shared__ float red2[256];
    const int row = blockIdx.x, tid = threadIdx.x;
    const float* pa = xa + (size_t)row * DD;
    const float* pb = xb + (size_t)row * DD;

    float v[4], s = 0.f, s2 = 0.f;
#pragma unroll
    for (int i = 0; i < 4; ++i) {
        float t = pa[tid + 256 * i] + pb[tid + 256 * i];
        v[i] = t; s += t; s2 += t * t;
    }
    red[tid] = s; red2[tid] = s2;
    for (int off = 128; off > 0; off >>= 1) {
        __syncthreads();
        if (tid < off) { red[tid] += red[tid + off]; red2[tid] += red2[tid + off]; }
    }
    __syncthreads();
    const float mu   = red[0] * (1.0f / DD);
    const float var  = red2[0] * (1.0f / DD) - mu * mu;
    const float rinv = rsqrtf(var + 1e-5f);
#pragma unroll
    for (int i = 0; i < 4; ++i) {
        const int c = tid + 256 * i;
        const float r = (v[i] - mu) * rinv * gamma[c] + beta[c];
        o32[(size_t)row * DD + c] = r;
        if (o16) o16[(size_t)row * DD + c] = f2bf(r);
    }
}

// ---------------------------------------------------------------------------
// Orchestration
// ---------------------------------------------------------------------------
extern "C" void kernel_launch(void* const* d_in, const int* in_sizes, int n_in,
                              void* d_out, int out_size, void* d_ws, size_t ws_size,
                              hipStream_t stream) {
    (void)in_sizes; (void)n_in; (void)out_size; (void)ws_size;

    const float* x   = (const float*)d_in[0];
    const float* wq  = (const float*)d_in[1];
    const float* wk  = (const float*)d_in[2];
    const float* wo  = (const float*)d_in[3];
    const float* g1  = (const float*)d_in[4];
    const float* b1  = (const float*)d_in[5];
    const float* w1  = (const float*)d_in[6];
    const float* bb1 = (const float*)d_in[7];
    const float* w2  = (const float*)d_in[8];
    const float* bb2 = (const float*)d_in[9];
    const float* g2  = (const float*)d_in[10];
    const float* b2  = (const float*)d_in[11];
    float* out = (float*)d_out;

    // workspace carve-up (bf16 region then fp32 region; ~126 MiB total)
    char* ws = (char*)d_ws;
    u16* x_bf  = (u16*)ws;                       // 4096*1024
    u16* wq_bf = x_bf  + (size_t)MM * DD;
    u16* wk_bf = wq_bf + (size_t)DD * DD;
    u16* wo_bf = wk_bf + (size_t)DD * DD;
    u16* w1_bf = wo_bf + (size_t)DD * DD;        // 1024*4096
    u16* w2_bf = w1_bf + (size_t)DD * FF;        // 4096*1024
    u16* qh    = w2_bf + (size_t)FF * DD;        // [b][n][l][hd]
    u16* kh    = qh    + (size_t)BB * NH * LL * HDIM;
    u16* ctxb  = kh    + (size_t)BB * NH * LL * HDIM;   // [b][l][d] bf16
    u16* h_bf  = ctxb  + (size_t)MM * DD;
    u16* ff1   = h_bf  + (size_t)MM * DD;        // 4096*4096 bf16
    float* tmp = (float*)(ff1 + (size_t)MM * FF);       // 4096*1024 f32
    float* hf  = tmp + (size_t)MM * DD;                 // 4096*1024 f32

    // 1. convert inputs/weights to bf16 (8 elems / thread)
    cvt_kernel<<<(MM * DD) / 2048, 256, 0, stream>>>(x,  x_bf,  MM * DD);
    cvt_kernel<<<(DD * DD) / 2048, 256, 0, stream>>>(wq, wq_bf, DD * DD);
    cvt_kernel<<<(DD * DD) / 2048, 256, 0, stream>>>(wk, wk_bf, DD * DD);
    cvt_kernel<<<(DD * DD) / 2048, 256, 0, stream>>>(wo, wo_bf, DD * DD);
    cvt_kernel<<<(DD * FF) / 2048, 256, 0, stream>>>(w1, w1_bf, DD * FF);
    cvt_kernel<<<(FF * DD) / 2048, 256, 0, stream>>>(w2, w2_bf, FF * DD);

    // 2/3. Q,K projections -> head layout (Q pre-scaled by 1/sqrt(64))
    dim3 blk(256);
    gemm_wmma_kernel<false, false, 2><<<dim3(DD / 128, MM / 128), blk, 0, stream>>>(
        x_bf, wq_bf, nullptr, qh, MM, DD, DD, 0.125f);
    gemm_wmma_kernel<false, false, 2><<<dim3(DD / 128, MM / 128), blk, 0, stream>>>(
        x_bf, wk_bf, nullptr, kh, MM, DD, DD, 1.0f);

    // 4. flash attention (attn contracted with K, per reference)
    flash_kernel<<<dim3(LL / 64, BB * NH), dim3(128), 0, stream>>>(qh, kh, ctxb);

    // 5. attn_out = ctx @ wo  (fp32 out)
    gemm_wmma_kernel<false, false, 0><<<dim3(DD / 128, MM / 128), blk, 0, stream>>>(
        ctxb, wo_bf, nullptr, tmp, MM, DD, DD, 1.0f);

    // 6. h = LN(x + attn_out)
    add_ln_kernel<<<MM, 256, 0, stream>>>(x, tmp, g1, b1, hf, h_bf);

    // 7. ff1 = gelu(h @ w1 + bb1)  (bf16 out)
    gemm_wmma_kernel<true, true, 1><<<dim3(FF / 128, MM / 128), blk, 0, stream>>>(
        h_bf, w1_bf, bb1, ff1, MM, DD, FF, 1.0f);

    // 8. ff2 = ff1 @ w2 + bb2  (fp32 out)
    gemm_wmma_kernel<true, false, 0><<<dim3(DD / 128, MM / 128), blk, 0, stream>>>(
        ff1, w2_bf, bb2, tmp, MM, FF, DD, 1.0f);

    // 9. out = LN(h + ff2)
    add_ln_kernel<<<MM, 256, 0, stream>>>(hf, tmp, g2, b2, out, nullptr);
}
